// TGAM_53652731462314
// MI455X (gfx1250) — matrix operand order repaired and checked by reading, not compiled, not run on hardware
//
#include <hip/hip_runtime.h>

typedef __bf16 bf16_t;
typedef __attribute__((ext_vector_type(16))) __bf16 v16bf;
typedef __attribute__((ext_vector_type(8)))  float  v8f;

#define WMMA_BF16(a, b, c) \
  __builtin_amdgcn_wmma_f32_16x16x32_bf16(false, (a), false, (b), (short)0, (c), false, false)

// ================= WMMA fragment layout helpers (CDNA5 wave32, ISA 7.12.2) =================
//
// LDS is pre-swizzled so every lane's 16 bf16 fragment elements are 32 contiguous
// bytes -> ds_load_b128 x2 per fragment instead of 16 scalar b16 loads + v_mov_b16 packs.

// B-matrix (Kx128 weights): element (k,n) -> fragment slot.
// b_frag semantics: col = n0 + lane&15 ; kb = k0 + 8*(lane>=16); elems 0..7 -> K=kb+0..7 ; elems 8..15 -> K=kb+16..23.
__device__ __forceinline__ int b_swz_idx(int k, int n) {
  int ktile = k >> 5, o = k & 31;
  int ntile = n >> 4, cc = n & 15;
  int lane = cc + ((o & 8) ? 16 : 0);
  int j = ((o & 16) ? 8 : 0) + (o & 7);
  return (((ktile * 8 + ntile) * 32) + lane) * 16 + j;
}

__device__ __forceinline__ v16bf b_frag_swz(const bf16_t* __restrict__ w, int ktile, int ntile) {
  int lane = threadIdx.x & 31;
  return *(const v16bf*)(w + (((ktile * 8 + ntile) * 32) + lane) * 16);
}

// A-matrix (16x128 hidden tile): element (r, k) -> fragment slot.
// a_frag semantics: row = lane&15 ; kb = k0 + 8*(lane>=16);
// elems 0..7 -> K=kb+0..7 ; elems 8..15 -> K=kb+16..23.
__device__ __forceinline__ int a_swz_idx(int r, int k) {
  int ktile = k >> 5, o = k & 31;
  int lane = r + ((o & 8) ? 16 : 0);
  int j = ((o & 16) ? 8 : 0) + (o & 7);
  return (ktile * 32 + lane) * 16 + j;
}

__device__ __forceinline__ v16bf a_frag_swz(const bf16_t* __restrict__ sh, int ktile) {
  int lane = threadIdx.x & 31;
  return *(const v16bf*)(sh + (ktile * 32 + lane) * 16);
}

// A fragment straight from a per-lane fp32 row pointer (row = lane&15).
__device__ __forceinline__ v16bf a_frag_f32(const float* __restrict__ row, int k0) {
  int lane = threadIdx.x & 31;
  int kb = k0 + ((lane >> 4) << 3);
  v16bf a;
#pragma unroll
  for (int j = 0; j < 8; ++j) a[j] = (bf16_t)row[kb + j];
#pragma unroll
  for (int j = 0; j < 8; ++j) a[8 + j] = (bf16_t)row[kb + 16 + j];
  return a;
}

// Same but A = r1 + r2 (fused "node_emb + acc" for the update MLP).
__device__ __forceinline__ v16bf a_frag_f32_sum(const float* __restrict__ r1,
                                                const float* __restrict__ r2, int k0) {
  int lane = threadIdx.x & 31;
  int kb = k0 + ((lane >> 4) << 3);
  v16bf a;
#pragma unroll
  for (int j = 0; j < 8; ++j) a[j] = (bf16_t)(r1[kb + j] + r2[kb + j]);
#pragma unroll
  for (int j = 0; j < 8; ++j) a[8 + j] = (bf16_t)(r1[kb + 16 + j] + r2[kb + 16 + j]);
  return a;
}

// Stage a Kx128 fp32 weight matrix into swizzled bf16 LDS.
template <int K>
__device__ __forceinline__ void stage_weights(bf16_t* __restrict__ sW,
                                              const float* __restrict__ W) {
  for (int i = threadIdx.x; i < K * 128; i += 256) {
    int k = i >> 7, n = i & 127;
    sW[b_swz_idx(k, n)] = (bf16_t)W[i];
  }
}

// Bias+ReLU the 8 C-tiles and write the 16x128 hidden tile into swizzled-A LDS.
__device__ __forceinline__ void store_hidden(bf16_t* __restrict__ myH, const v8f* c,
                                             const float* __restrict__ B1) {
  int lane = threadIdx.x & 31;
  int rbase = (lane >> 4) << 3;
#pragma unroll
  for (int n = 0; n < 8; ++n) {
    int col = n * 16 + (lane & 15);
    float bias = B1[col];
#pragma unroll
    for (int i = 0; i < 8; ++i) {
      float v = c[n][i] + bias;
      myH[a_swz_idx(rbase + i, col)] = (bf16_t)(v > 0.f ? v : 0.f);
    }
  }
}

// Second layer: hidden(16x128, swizzled LDS) @ W2(128x128, swizzled LDS).
__device__ __forceinline__ void layer2(v8f* d, const bf16_t* __restrict__ myH,
                                       const bf16_t* __restrict__ sW) {
#pragma unroll
  for (int kt = 0; kt < 4; ++kt) {
    v16bf a = a_frag_swz(myH, kt);
#pragma unroll
    for (int n = 0; n < 8; ++n) d[n] = WMMA_BF16(a, b_frag_swz(sW, kt, n), d[n]);
  }
}

// ================= generic 2-layer MLP:  out = relu(X@W1+b1)@W2+b2 =================
// 256 threads = 8 waves, 16 rows/wave -> 128 rows/block.
template <int K1>
__global__ __launch_bounds__(256) void mlp2_kernel(
    const float* __restrict__ X, int M,
    const float* __restrict__ W1, const float* __restrict__ B1,
    const float* __restrict__ W2, const float* __restrict__ B2,
    float* __restrict__ out) {
  __shared__ bf16_t sW[128 * 128];
  __shared__ bf16_t sH[128 * 128];

  stage_weights<K1>(sW, W1);
  __syncthreads();

  int lane = threadIdx.x & 31, wv = threadIdx.x >> 5;
  int m0 = blockIdx.x * 128 + wv * 16;
  int mr = m0 + (lane & 15);
  int mc = mr < M ? mr : M - 1;
  const float* xrow = X + (size_t)mc * K1;

  v8f c[8] = {};
#pragma unroll
  for (int kt = 0; kt < K1 / 32; ++kt) {
    v16bf a = a_frag_f32(xrow, kt * 32);
#pragma unroll
    for (int n = 0; n < 8; ++n) c[n] = WMMA_BF16(a, b_frag_swz(sW, kt, n), c[n]);
  }

  bf16_t* myH = sH + wv * 2048;
  store_hidden(myH, c, B1);
  __syncthreads();
  stage_weights<128>(sW, W2);
  __syncthreads();

  v8f d[8] = {};
  layer2(d, myH, sW);

  int rbase = (lane >> 4) << 3;
#pragma unroll
  for (int n = 0; n < 8; ++n) {
    int col = n * 16 + (lane & 15);
    float bias = B2[col];
#pragma unroll
    for (int i = 0; i < 8; ++i) {
      int r = m0 + rbase + i;
      if (r < M) out[(size_t)r * 128 + col] = d[n][i] + bias;
    }
  }
}

// ================= node update: out = relu((emb+acc)@W1+b1)@W2+b2 =================
__global__ __launch_bounds__(256) void update_kernel(
    const float* __restrict__ emb, const float* __restrict__ acc, int M,
    const float* __restrict__ W1, const float* __restrict__ B1,
    const float* __restrict__ W2, const float* __restrict__ B2,
    float* __restrict__ out) {
  __shared__ bf16_t sW[128 * 128];
  __shared__ bf16_t sH[128 * 128];

  stage_weights<128>(sW, W1);
  __syncthreads();

  int lane = threadIdx.x & 31, wv = threadIdx.x >> 5;
  int m0 = blockIdx.x * 128 + wv * 16;
  int mr = m0 + (lane & 15);
  int mc = mr < M ? mr : M - 1;
  const float* r1 = emb + (size_t)mc * 128;
  const float* r2 = acc + (size_t)mc * 128;

  v8f c[8] = {};
#pragma unroll
  for (int kt = 0; kt < 4; ++kt) {
    v16bf a = a_frag_f32_sum(r1, r2, kt * 32);
#pragma unroll
    for (int n = 0; n < 8; ++n) c[n] = WMMA_BF16(a, b_frag_swz(sW, kt, n), c[n]);
  }

  bf16_t* myH = sH + wv * 2048;
  store_hidden(myH, c, B1);
  __syncthreads();
  stage_weights<128>(sW, W2);
  __syncthreads();

  v8f d[8] = {};
  layer2(d, myH, sW);

  int rbase = (lane >> 4) << 3;
#pragma unroll
  for (int n = 0; n < 8; ++n) {
    int col = n * 16 + (lane & 15);
    float bias = B2[col];
#pragma unroll
    for (int i = 0; i < 8; ++i) {
      int r = m0 + rbase + i;
      if (r < M) out[(size_t)r * 128 + col] = d[n][i] + bias;
    }
  }
}

// ================= message kernel =================
// m = relu(concat[h_a, ee, h_b] @ W1(384x128) + b1) @ W2(128x128) + b2 ; acc[el[io]] += m
__global__ __launch_bounds__(256) void msg_kernel(
    const float* __restrict__ emb, const float* __restrict__ ee,
    const int* __restrict__ el, int E, int ia, int ib, int io,
    const float* __restrict__ W1, const float* __restrict__ B1,
    const float* __restrict__ W2, const float* __restrict__ B2,
    float* __restrict__ acc) {
  __shared__ bf16_t sW[384 * 128];  // 96 KB, reused for W2 (32 KB) in layer 2
  __shared__ bf16_t sH[128 * 128];  // 32 KB

  stage_weights<384>(sW, W1);
  __syncthreads();

  int lane = threadIdx.x & 31, wv = threadIdx.x >> 5;
  int e0 = blockIdx.x * 128 + wv * 16;
  int me = e0 + (lane & 15);
  int ec = me < E ? me : E - 1;
  const float* pA = emb + (size_t)el[2 * ec + ia] * 128;
  const float* pE = ee  + (size_t)ec * 128;
  const float* pB = emb + (size_t)el[2 * ec + ib] * 128;

  v8f c[8] = {};
#pragma unroll
  for (int kt = 0; kt < 4; ++kt) {
    v16bf a = a_frag_f32(pA, kt * 32);
#pragma unroll
    for (int n = 0; n < 8; ++n) c[n] = WMMA_BF16(a, b_frag_swz(sW, kt, n), c[n]);
  }
#pragma unroll
  for (int kt = 0; kt < 4; ++kt) {
    v16bf a = a_frag_f32(pE, kt * 32);
#pragma unroll
    for (int n = 0; n < 8; ++n) c[n] = WMMA_BF16(a, b_frag_swz(sW, 4 + kt, n), c[n]);
  }
#pragma unroll
  for (int kt = 0; kt < 4; ++kt) {
    v16bf a = a_frag_f32(pB, kt * 32);
#pragma unroll
    for (int n = 0; n < 8; ++n) c[n] = WMMA_BF16(a, b_frag_swz(sW, 8 + kt, n), c[n]);
  }

  bf16_t* myH = sH + wv * 2048;
  store_hidden(myH, c, B1);
  __syncthreads();
  stage_weights<128>(sW, W2);
  __syncthreads();

  v8f d[8] = {};
  layer2(d, myH, sW);

  // scatter-add into acc via hardware f32 atomics
  int rbase = (lane >> 4) << 3;
#pragma unroll
  for (int n = 0; n < 8; ++n) {
    int col = n * 16 + (lane & 15);
    float bias = B2[col];
#pragma unroll
    for (int i = 0; i < 8; ++i) {
      int e = e0 + rbase + i;
      if (e < E) {
        int node = el[2 * e + io];
        unsafeAtomicAdd(&acc[(size_t)node * 128 + col], d[n][i] + bias);
      }
    }
  }
}

// ================= host orchestration =================
extern "C" void kernel_launch(void* const* d_in, const int* in_sizes, int n_in,
                              void* d_out, int out_size, void* d_ws, size_t ws_size,
                              hipStream_t stream) {
  const float* node_features = (const float*)d_in[0];
  const int*   edge_list     = (const int*)d_in[1];
  const float* edge_features = (const float*)d_in[2];
  const float* nw1 = (const float*)d_in[4],  *nb1 = (const float*)d_in[5];
  const float* nw2 = (const float*)d_in[6],  *nb2 = (const float*)d_in[7];
  const float* ew1 = (const float*)d_in[8],  *eb1 = (const float*)d_in[9];
  const float* ew2 = (const float*)d_in[10], *eb2 = (const float*)d_in[11];
  const float* mw1[2] = {(const float*)d_in[12], (const float*)d_in[16]};
  const float* mb1[2] = {(const float*)d_in[13], (const float*)d_in[17]};
  const float* mw2[2] = {(const float*)d_in[14], (const float*)d_in[18]};
  const float* mb2[2] = {(const float*)d_in[15], (const float*)d_in[19]};
  const float* aw1 = (const float*)d_in[20], *ab1 = (const float*)d_in[21];
  const float* aw2 = (const float*)d_in[22], *ab2 = (const float*)d_in[23];

  int N = in_sizes[0] / 128;
  int E = in_sizes[1] / 2;

  float* embA = (float*)d_ws;                 // N x 128
  float* embB = embA + (size_t)N * 128;       // N x 128
  float* acc  = embB + (size_t)N * 128;       // N x 128
  float* ee   = acc  + (size_t)N * 128;       // E x 128

  dim3 blk(256);
  int nodeBlocks = (N + 127) / 128;
  int edgeBlocks = (E + 127) / 128;

  mlp2_kernel<128><<<nodeBlocks, blk, 0, stream>>>(node_features, N, nw1, nb1, nw2, nb2, embA);
  mlp2_kernel<64><<<edgeBlocks, blk, 0, stream>>>(edge_features, E, ew1, eb1, ew2, eb2, ee);

  const float* cur = embA;
  for (int r = 0; r < 2; ++r) {
    hipMemsetAsync(acc, 0, (size_t)N * 128 * sizeof(float), stream);
    // forward: concat[h_src, ee, h_dst] -> scatter to dst
    msg_kernel<<<edgeBlocks, blk, 0, stream>>>(cur, ee, edge_list, E, 0, 1, 1,
                                               mw1[r], mb1[r], mw2[r], mb2[r], acc);
    // backward: concat[h_dst, ee, h_src] -> scatter to src
    msg_kernel<<<edgeBlocks, blk, 0, stream>>>(cur, ee, edge_list, E, 1, 0, 0,
                                               mw1[r], mb1[r], mw2[r], mb2[r], acc);
    float* dst = (r == 0) ? embB : (float*)d_out;
    update_kernel<<<nodeBlocks, blk, 0, stream>>>(cur, acc, N, aw1, ab1, aw2, ab2, dst);
    cur = dst;
  }
}
